// DAttentionBaseline_31954556682602
// MI455X (gfx1250) — matrix operand order repaired and checked by reading, not compiled, not run on hardware
//
#include <hip/hip_runtime.h>
#include <hip/hip_bf16.h>
#include <math.h>
#include <stdint.h>

typedef __attribute__((ext_vector_type(16))) __bf16 v16bf;
typedef __attribute__((ext_vector_type(8)))  float  v8f;
typedef unsigned int uint32x4 __attribute__((ext_vector_type(4)));
typedef int          int32x4  __attribute__((ext_vector_type(4)));
typedef int          int32x8  __attribute__((ext_vector_type(8)));

#define WMMA_BF16(a, b, c) \
  __builtin_amdgcn_wmma_f32_16x16x32_bf16(false, (a), false, (b), (short)0, (c), false, false)

// ---------------------------------------------------------------------------
// Fragment loaders (wave32 WMMA; this file's contiguous k order, k = 16*(lane/16) + element, same for A and B)
// A 16x32 bf16: lanes 0-15: M=lane, K=0..15; lanes 16-31: M=lane-16,
//               K=16..31; elements packed 2/VGPR in order.
// B 32x16 bf16: lanes 0-15: N=lane, K=0..15; lanes 16-31: N=lane-16, K=16..31.
// ---------------------------------------------------------------------------

// A[m,k] = A[m*lda + k] (row-major fp32 source)
__device__ inline v16bf load_a_rowmajor(const float* __restrict__ A, int lda, int lane) {
  int m  = lane & 15;
  int kb = (lane < 16) ? 0 : 16;
  const float* r = A + (size_t)m * lda;
  v16bf a;
#pragma unroll
  for (int j = 0; j < 8; ++j) a[j]     = (__bf16)r[kb + j];
#pragma unroll
  for (int j = 0; j < 8; ++j) a[8 + j] = (__bf16)r[kb + 8 + j];
  return a;
}

// A[m,k] = A[k*ld + m] (k-major fp32 source, i.e. A = Xᵀ)
__device__ inline v16bf load_a_kmajor(const float* __restrict__ A, int ld, int lane) {
  int m  = lane & 15;
  int kb = (lane < 16) ? 0 : 16;
  v16bf a;
#pragma unroll
  for (int j = 0; j < 8; ++j) a[j]     = (__bf16)A[(size_t)(kb + j) * ld + m];
#pragma unroll
  for (int j = 0; j < 8; ++j) a[8 + j] = (__bf16)A[(size_t)(kb + 8 + j) * ld + m];
  return a;
}

// A row-major, bf16 source in LDS
__device__ inline v16bf load_a_bf16(const __bf16* __restrict__ A, int lda, int lane) {
  int m  = lane & 15;
  int kb = (lane < 16) ? 0 : 16;
  const __bf16* r = A + (size_t)m * lda;
  v16bf a;
#pragma unroll
  for (int j = 0; j < 8; ++j) a[j]     = r[kb + j];
#pragma unroll
  for (int j = 0; j < 8; ++j) a[8 + j] = r[kb + 8 + j];
  return a;
}

// B[k,n] = B[k*ldb + n]  (row-major fp32 source)
__device__ inline v16bf load_b_rowmajor(const float* __restrict__ B, int ldb, int lane) {
  int n  = lane & 15;
  int kb = (lane < 16) ? 0 : 16;
  v16bf b;
#pragma unroll
  for (int j = 0; j < 16; ++j) b[j] = (__bf16)B[(size_t)(kb + j) * ldb + n];
  return b;
}

// B[k,n] = B[k*ldb + n]  (row-major bf16 source in LDS)
__device__ inline v16bf load_b_rowmajor_lds(const __bf16* __restrict__ B, int ldb, int lane) {
  int n  = lane & 15;
  int kb = (lane < 16) ? 0 : 16;
  v16bf b;
#pragma unroll
  for (int j = 0; j < 16; ++j) b[j] = B[(size_t)(kb + j) * ldb + n];
  return b;
}

// B[k,n] = B[n*ld + k] (k-contiguous bf16 source in LDS, i.e. B = Vᵀ)
__device__ inline v16bf load_b_kcontig_lds(const __bf16* __restrict__ B, int ld, int lane) {
  int n  = lane & 15;
  int kb = (lane < 16) ? 0 : 16;
  const __bf16* r = B + (size_t)n * ld + kb;
  v16bf b;
#pragma unroll
  for (int j = 0; j < 16; ++j) b[j] = r[j];
  return b;
}

// Bilinear sample (pixel coords), zeros padding outside [0,W-1]x[0,H-1]
__device__ inline float bilin(const float* __restrict__ img, int W, int H, float fx, float fy) {
  float x0f = floorf(fx), y0f = floorf(fy);
  float wx1 = fx - x0f, wy1 = fy - y0f;
  float wx0 = 1.f - wx1, wy0 = 1.f - wy1;
  int x0 = (int)x0f, y0 = (int)y0f;
  int x1 = x0 + 1,  y1 = y0 + 1;
  float acc = 0.f;
  if (x0 >= 0 && x0 < W) {
    if (y0 >= 0 && y0 < H) acc += img[y0 * W + x0] * (wx0 * wy0);
    if (y1 >= 0 && y1 < H) acc += img[y1 * W + x0] * (wx0 * wy1);
  }
  if (x1 >= 0 && x1 < W) {
    if (y0 >= 0 && y0 < H) acc += img[y0 * W + x1] * (wx1 * wy0);
    if (y1 >= 0 && y1 < H) acc += img[y1 * W + x1] * (wx1 * wy1);
  }
  return acc;
}

// ---------------------------------------------------------------------------
// 1x1-conv GEMM: OUT[b,m,p] = sum_k W[m,k]*IN[b,k,p] + bias[m]
// block = 128 (4 waves); each wave owns a 16(M) x 64(N) strip: A fragment
// loaded once per k-step, reused across 4 WMMAs. grid (N/256, M/16, B)
// ---------------------------------------------------------------------------
__global__ void gemm_proj_kernel(const float* __restrict__ W, const float* __restrict__ bias,
                                 const float* __restrict__ IN, float* __restrict__ OUT,
                                 int M, int N, int K) {
  int wave = threadIdx.x >> 5;
  int lane = threadIdx.x & 31;
  int n0 = (blockIdx.x * 4 + wave) * 64;
  int m0 = blockIdx.y * 16;
  int b  = blockIdx.z;
  const float* Wt = W + (size_t)m0 * K;
  const float* Bt = IN + (size_t)b * K * N + n0;
  v8f acc[4];
#pragma unroll
  for (int t = 0; t < 4; ++t) acc[t] = v8f{};
  for (int k0 = 0; k0 < K; k0 += 32) {
    __builtin_prefetch(Wt + k0 + 32, 0, 0);                       // global_prefetch_b8
    __builtin_prefetch(Bt + (size_t)(k0 + 32) * N, 0, 0);
    v16bf a = load_a_rowmajor(Wt + k0, K, lane);
#pragma unroll
    for (int t = 0; t < 4; ++t) {
      v16bf bb = load_b_rowmajor(Bt + (size_t)k0 * N + t * 16, N, lane);
      acc[t] = WMMA_BF16(a, bb, acc[t]);
    }
  }
  int nl    = lane & 15;
  int mbase = (lane < 16) ? 0 : 8;
  float* Ct = OUT + (size_t)b * M * N + (size_t)m0 * N + n0;
#pragma unroll
  for (int t = 0; t < 4; ++t)
#pragma unroll
    for (int r = 0; r < 8; ++r) {
      int m = mbase + r;
      Ct[(size_t)m * N + t * 16 + nl] = acc[t][r] + bias[m0 + m];
    }
}

// ---------------------------------------------------------------------------
// offset predictor: depthwise3x3/s2/p1 -> LN(channel) -> GELU -> 1x1 -> pos
// block = 128 (one thread per group-channel), grid (Wk=16, Hk=16, B*g=64)
// ---------------------------------------------------------------------------
__global__ void offset_kernel(const float* __restrict__ Q, const float* __restrict__ dw_w,
                              const float* __restrict__ dw_b, const float* __restrict__ ln_w,
                              const float* __restrict__ ln_b, const float* __restrict__ pw_w,
                              float* __restrict__ POS, float* __restrict__ pos_out,
                              float* __restrict__ ref_out) {
  __shared__ float buf[128];
  int c  = threadIdx.x;
  int wk = blockIdx.x, hk = blockIdx.y, bg = blockIdx.z;
  int b = bg >> 2, gi = bg & 3;
  const float* qb = Q + ((size_t)(b * 512 + gi * 128 + c)) * 1024;
  float s = dw_b[c];
#pragma unroll
  for (int r = 0; r < 3; ++r) {
    int iy = 2 * hk - 1 + r;
    if (iy < 0 || iy > 31) continue;
#pragma unroll
    for (int t = 0; t < 3; ++t) {
      int ix = 2 * wk - 1 + t;
      if (ix < 0 || ix > 31) continue;
      s += dw_w[c * 9 + r * 3 + t] * qb[iy * 32 + ix];
    }
  }
  buf[c] = s; __syncthreads();
  for (int st = 64; st > 0; st >>= 1) { if (c < st) buf[c] += buf[c + st]; __syncthreads(); }
  float mu = buf[0] * (1.f / 128.f); __syncthreads();
  float d = s - mu;
  buf[c] = d * d; __syncthreads();
  for (int st = 64; st > 0; st >>= 1) { if (c < st) buf[c] += buf[c + st]; __syncthreads(); }
  float var = buf[0] * (1.f / 128.f); __syncthreads();
  float t = d * rsqrtf(var + 1e-5f) * ln_w[c] + ln_b[c];
  t = 0.5f * t * (1.f + erff(t * 0.70710678118654752f));   // exact GELU
  buf[c] = t * pw_w[c]; __syncthreads();
  for (int st = 64; st > 0; st >>= 1) { if (c < st) buf[c] += buf[c + st]; __syncthreads(); }
  float oy = buf[0]; __syncthreads();
  buf[c] = t * pw_w[128 + c]; __syncthreads();
  for (int st = 64; st > 0; st >>= 1) { if (c < st) buf[c] += buf[c + st]; __syncthreads(); }
  float ox = buf[0];
  if (c == 0) {
    float ry = (hk + 0.5f) * (2.f / 15.f) - 1.f;
    float rx = (wk + 0.5f) * (2.f / 15.f) - 1.f;
    float py = fminf(1.f, fmaxf(-1.f, oy + ry));
    float px = fminf(1.f, fmaxf(-1.f, ox + rx));
    size_t pi = ((size_t)bg * 256 + hk * 16 + wk) * 2;
    POS[pi] = py;     POS[pi + 1] = px;
    pos_out[pi] = py; pos_out[pi + 1] = px;
    ref_out[pi] = ry; ref_out[pi + 1] = rx;
  }
}

// ---------------------------------------------------------------------------
// grid_sample of y at pos -> y_s (B, C, ns)
// ---------------------------------------------------------------------------
__global__ void sample_y_kernel(const float* __restrict__ Y, const float* __restrict__ POS,
                                float* __restrict__ YS) {
  int idx = blockIdx.x * 256 + threadIdx.x;   // 64 * 128 * 256 = 2,097,152
  int n  = idx & 255;
  int cc = (idx >> 8) & 127;
  int bg = idx >> 15;
  size_t pi = ((size_t)bg * 256 + n) * 2;
  float py = POS[pi], px = POS[pi + 1];
  int b = bg >> 2, gi = bg & 3, ch = gi * 128 + cc;
  const float* img = Y + ((size_t)b * 512 + ch) * 1024;
  float fx = (px + 1.f) * 0.5f * 31.f;
  float fy = (py + 1.f) * 0.5f * 31.f;
  YS[((size_t)b * 512 + ch) * 256 + n] = bilin(img, 32, 32, fx, fy);
}

// ---------------------------------------------------------------------------
// fused attention. Per block (128 thr / 4 waves): K,V panels (64x256) staged
// once in LDS as bf16 (shared by all 4 waves); rpe 63x63 table staged via the
// Tensor Data Mover (TDM) when available; each wave owns a 16-row tile whose
// 16x256 logits live in accumulator fragments; bias = bilinear from LDS rpe;
// softmax via __shfl_xor; probs -> LDS bf16 -> WMMA against V.
// grid (HW/64 = 16, B*nh = 128)
// ---------------------------------------------------------------------------
__global__ void attn_kernel(const float* __restrict__ Q, const float* __restrict__ K,
                            const float* __restrict__ V, const float* __restrict__ POS,
                            const float* __restrict__ RPE, float* __restrict__ AO) {
  __shared__ float  s_rpe[63 * 63];          // 15.9 KB
  __shared__ float  s_pos[256 * 2];          //  2.0 KB
  __shared__ __bf16 s_K[64 * 256];           // 32.0 KB
  __shared__ __bf16 s_V[64 * 256];           // 32.0 KB
  __shared__ __bf16 s_P[4][16][256];         // 32.0 KB   (~114 KB total, <320KB WGP)

  int wave = threadIdx.x >> 5;
  int lane = threadIdx.x & 31;
  int bh = blockIdx.y;
  int b = bh >> 3, h = bh & 7;
  int bg = b * 4 + (h >> 1);
  int m0 = (blockIdx.x * 4 + wave) * 16;

  const float* rpe_h = RPE + (size_t)h * 63 * 63;
  const float* kbase = K + ((size_t)b * 512 + h * 64) * 256;   // k[c,n], row-major 64x256
  const float* vbase = V + ((size_t)b * 512 + h * 64) * 256;   // v[c,n]

  // ---- stage rpe via TDM (wave 0 only; done-signal tracked by TENSORcnt)
#if defined(__has_builtin) && __has_builtin(__builtin_amdgcn_tensor_load_to_lds)
  if (wave == 0) {
    uint64_t ga  = (uint64_t)(uintptr_t)rpe_h;
    uint32_t lds = (uint32_t)(uintptr_t)&s_rpe[0];   // flat addr[31:0] == LDS offset
    uint32x4 g0;
    g0[0] = 1u;                                      // count=1, user descriptor
    g0[1] = lds;                                     // lds_addr
    g0[2] = (uint32_t)ga;                            // global_addr[31:0]
    g0[3] = (uint32_t)((ga >> 32) & 0x01FFFFFFu) | (2u << 30);  // addr[56:32] | type=2
    int32x8 g1;
    g1[0] = (int)(2u << 16);                         // data_size = 4 bytes
    g1[1] = (int)((3969u & 0xFFFFu) << 16);          // tensor_dim0[15:0] @ bits 63:48
    g1[2] = (int)((3969u >> 16) | (1u << 16));       // tensor_dim0[31:16] | tensor_dim1[15:0]
    g1[3] = (int)(3969u << 16);                      // tensor_dim1[31:16]=0 | tile_dim0
    g1[4] = (int)1;                                  // tile_dim1 = 1 (single row of tile)
    g1[5] = (int)3969;                               // tensor_dim0_stride[31:0]
    g1[6] = 0;
    g1[7] = 0;
    int32x4 z4 = {};
#if __clang_major__ >= 23
    int32x8 z8 = {};
    __builtin_amdgcn_tensor_load_to_lds(g0, g1, z4, z4, z8, 0);
#else
    __builtin_amdgcn_tensor_load_to_lds(g0, g1, z4, z4, 0);
#endif
    __builtin_amdgcn_s_wait_tensorcnt(0);
  }
#else
  for (int i = threadIdx.x; i < 63 * 63; i += 128) s_rpe[i] = rpe_h[i];
#endif

  // ---- cooperative staging: pos + bf16 K/V panels (shared by all 4 waves)
  for (int i = threadIdx.x; i < 512; i += 128) s_pos[i] = POS[(size_t)bg * 512 + i];
  for (int i = threadIdx.x; i < 64 * 256; i += 128) {
    s_K[i] = (__bf16)kbase[i];
    s_V[i] = (__bf16)vbase[i];
  }
  __syncthreads();

  // ---- GEMM1: logits = (qf^T k) * scale, tile 16(HW) x 256(ns), K=hc=64
  const float* qbase = Q + ((size_t)b * 512 + h * 64) * 1024 + m0;  // qf[c,m]
  v16bf qa0 = load_a_kmajor(qbase, 1024, lane);                     // K = 0..31
  v16bf qa1 = load_a_kmajor(qbase + (size_t)32 * 1024, 1024, lane); // K = 32..63

  v8f acc[16];
#pragma unroll
  for (int t = 0; t < 16; ++t) acc[t] = v8f{};
#pragma unroll
  for (int t = 0; t < 16; ++t) {
    const __bf16* bp = s_K + t * 16;
    v16bf b0 = load_b_rowmajor_lds(bp, 256, lane);                  // k rows 0..31
    acc[t] = WMMA_BF16(qa0, b0, acc[t]);
    v16bf b1 = load_b_rowmajor_lds(bp + 32 * 256, 256, lane);       // k rows 32..63
    acc[t] = WMMA_BF16(qa1, b1, acc[t]);
  }

  // ---- bias + softmax (row-wise over ns=256)
  int nl    = lane & 15;
  int mbase = (lane < 16) ? 0 : 8;
#pragma unroll
  for (int r = 0; r < 8; ++r) {
    int m = m0 + mbase + r;
    float qy = (float)(m >> 5) * (2.f / 31.f) - 1.f;
    float qx = (float)(m & 31) * (2.f / 31.f) - 1.f;
    float mx = -1e30f;
#pragma unroll
    for (int t = 0; t < 16; ++t) {
      int n = t * 16 + nl;
      float py = s_pos[2 * n], px = s_pos[2 * n + 1];
      float fy = ((qy - py) * 0.5f + 1.f) * 31.f;   // (disp+1)*0.5*(63-1)
      float fx = ((qx - px) * 0.5f + 1.f) * 31.f;
      float l = acc[t][r] * 0.125f + bilin(s_rpe, 63, 63, fx, fy);
      acc[t][r] = l;
      mx = fmaxf(mx, l);
    }
#pragma unroll
    for (int d = 1; d <= 8; d <<= 1) mx = fmaxf(mx, __shfl_xor(mx, d, 32));
    float sum = 0.f;
#pragma unroll
    for (int t = 0; t < 16; ++t) {
      float e = __expf(acc[t][r] - mx);
      acc[t][r] = e;
      sum += e;
    }
#pragma unroll
    for (int d = 1; d <= 8; d <<= 1) sum += __shfl_xor(sum, d, 32);
    float inv = 1.f / sum;
#pragma unroll
    for (int t = 0; t < 16; ++t)
      s_P[wave][mbase + r][t * 16 + nl] = (__bf16)(acc[t][r] * inv);
  }

  // ---- GEMM2: out = P @ V^T, tile 16(HW) x 64(hc), K = ns = 256
  float* obase = AO + ((size_t)b * 512 + h * 64) * 1024;
  const __bf16* Pw = &s_P[wave][0][0];
#pragma unroll
  for (int ct = 0; ct < 4; ++ct) {
    v8f oacc = {};
#pragma unroll
    for (int ks = 0; ks < 8; ++ks) {
      v16bf pa = load_a_bf16(Pw + ks * 32, 256, lane);
      v16bf vb = load_b_kcontig_lds(s_V + (size_t)(ct * 16) * 256 + ks * 32, 256, lane);
      oacc = WMMA_BF16(pa, vb, oacc);
    }
#pragma unroll
    for (int r = 0; r < 8; ++r)
      obase[(size_t)(ct * 16 + nl) * 1024 + (m0 + mbase + r)] = oacc[r];
  }
}

// ---------------------------------------------------------------------------
// Workspace layout (floats):
//   q 0..8388608 | pos +32768 | y_s +2097152 | k +2097152 | v +2097152 | ao +8388608
// total 23,101,440 floats (~92.4 MB)
// ---------------------------------------------------------------------------
extern "C" void kernel_launch(void* const* d_in, const int* in_sizes, int n_in,
                              void* d_out, int out_size, void* d_ws, size_t ws_size,
                              hipStream_t stream) {
  (void)in_sizes; (void)n_in; (void)out_size; (void)ws_size;
  const float* x    = (const float*)d_in[0];
  const float* y    = (const float*)d_in[1];
  const float* dw_w = (const float*)d_in[2];
  const float* dw_b = (const float*)d_in[3];
  const float* ln_w = (const float*)d_in[4];
  const float* ln_b = (const float*)d_in[5];
  const float* pw_w = (const float*)d_in[6];
  const float* pq_w = (const float*)d_in[7];
  const float* pq_b = (const float*)d_in[8];
  const float* pk_w = (const float*)d_in[9];
  const float* pk_b = (const float*)d_in[10];
  const float* pv_w = (const float*)d_in[11];
  const float* pv_b = (const float*)d_in[12];
  const float* po_w = (const float*)d_in[13];
  const float* po_b = (const float*)d_in[14];
  const float* rpe  = (const float*)d_in[15];

  float* out = (float*)d_out;
  float* ws  = (float*)d_ws;
  float* q   = ws;
  float* pos = ws + 8388608;
  float* ys  = ws + 8421376;
  float* kk  = ws + 10518528;
  float* vv  = ws + 12615680;
  float* ao  = ws + 14712832;
  float* pos_out = out + 8388608;
  float* ref_out = out + 8421376;

  gemm_proj_kernel<<<dim3(4, 32, 16), 128, 0, stream>>>(pq_w, pq_b, x, q, 512, 1024, 512);
  offset_kernel<<<dim3(16, 16, 64), 128, 0, stream>>>(q, dw_w, dw_b, ln_w, ln_b, pw_w,
                                                      pos, pos_out, ref_out);
  sample_y_kernel<<<dim3(8192), 256, 0, stream>>>(y, pos, ys);
  gemm_proj_kernel<<<dim3(1, 32, 16), 128, 0, stream>>>(pk_w, pk_b, ys, kk, 512, 256, 512);
  gemm_proj_kernel<<<dim3(1, 32, 16), 128, 0, stream>>>(pv_w, pv_b, ys, vv, 512, 256, 512);
  attn_kernel<<<dim3(16, 128), 128, 0, stream>>>(q, kk, vv, pos, rpe, ao);
  gemm_proj_kernel<<<dim3(4, 32, 16), 128, 0, stream>>>(po_w, po_b, ao, out, 512, 1024, 512);
}
